// SpeakingEncoder_18580028523004
// MI455X (gfx1250) — matrix operand order repaired and checked by reading, not compiled, not run on hardware
//
#include <hip/hip_runtime.h>
#include <hip/hip_bf16.h>

typedef __attribute__((ext_vector_type(16))) __bf16 v16bf;
typedef __attribute__((ext_vector_type(8)))  float  v8f;
typedef __attribute__((ext_vector_type(4)))  __bf16 bf4;
typedef int v4i __attribute__((vector_size(16)));        // pointee type of async-LDS builtin params
typedef unsigned uv4 __attribute__((vector_size(16)));   // plain 16B vector for LDS frag loads
typedef __attribute__((address_space(3))) const uv4 lds_uv4;  // forces ds_load_b128

#define BB 4
#define TT 2048
#define DD 1024
#define KCODES 8192
#define MROWS (BB * TT)          // 8192 rows of x
#define NSPLIT 8                 // K-dim split across workgroups (argmin merged via atomic u64 min)
#define NB_COLS (KCODES / NSPLIT)
#define MBLK 64                  // rows per workgroup
#define NBLK 128                 // codes per inner tile
#define KSTEPS (DD / 32)         // 32 bf16-WMMA k-steps

// LDS layout: A panel (64 x 1024 bf16, row padded +16B for bank-conflict-free b128)
// + double-buffered B tile for async global->LDS pipelining.
#define A_ROW_BYTES (DD * 2 + 16)        // 2064
#define A_BYTES (MBLK * A_ROW_BYTES)     // 132096
#define B_COL_BYTES 80                   // 64B of k-data + 16B pad (stride 20 dwords, conflict-free)
#define B_BUF_BYTES (NBLK * B_COL_BYTES) // 10240
#define SMEM_BYTES (A_BYTES + 2 * B_BUF_BYTES)  // 152576 < 320KB/WGP

// ---- CDNA5 async global->LDS copy (ASYNCcnt-tracked, no VGPR round trip) ----
#if defined(__has_builtin)
#if __has_builtin(__builtin_amdgcn_global_load_async_to_lds_b128)
#define USE_ASYNC_LDS 1
#endif
#endif

__device__ __forceinline__ void cp16_async(const void* g, void* l) {
#ifdef USE_ASYNC_LDS
  __builtin_amdgcn_global_load_async_to_lds_b128(
      (__attribute__((address_space(1))) v4i*)(void*)g,
      (__attribute__((address_space(3))) v4i*)l, /*offset=*/0, /*cpol=*/0);
#else
  *(uv4*)l = *(const uv4*)g;   // fallback: sync VGPR round trip
#endif
}

__device__ __forceinline__ void wait_async_all() {
#ifdef USE_ASYNC_LDS
  asm volatile("s_wait_asynccnt 0x0" ::: "memory");
#endif
}

__device__ __forceinline__ uv4 lds_read16(const unsigned char* p) {
  return *(lds_uv4*)p;   // explicit addrspace(3) load -> ds_load_b128
}

__global__ void init_keys_kernel(unsigned long long* keys) {
  int i = blockIdx.x * blockDim.x + threadIdx.x;
  if (i < MROWS) keys[i] = 0xFFFFFFFFFFFFFFFFull;
}

__global__ __launch_bounds__(256) void convert_x_kernel(const float* __restrict__ x,
                                                        __bf16* __restrict__ xb) {
  int i = blockIdx.x * 256 + threadIdx.x;           // one float4 per thread
  float4 v = ((const float4*)x)[i];
  bf4 o;
  o.x = (__bf16)v.x; o.y = (__bf16)v.y; o.z = (__bf16)v.z; o.w = (__bf16)v.w;
  ((bf4*)xb)[i] = o;
}

// Convert codebook to bf16 and compute ||c_k||^2 (f32, full precision). One wave per code row.
__global__ __launch_bounds__(256) void convert_cb_kernel(const float* __restrict__ cb,
                                                         __bf16* __restrict__ cbb,
                                                         float* __restrict__ csq) {
  int wave = threadIdx.x >> 5;
  int lane = threadIdx.x & 31;
  int row = blockIdx.x * 8 + wave;
  const float4* src = (const float4*)(cb + (size_t)row * DD);
  bf4* dst = (bf4*)(cbb + (size_t)row * DD);
  float s = 0.0f;
#pragma unroll
  for (int i = 0; i < 8; ++i) {
    float4 v = src[i * 32 + lane];
    s = fmaf(v.x, v.x, s); s = fmaf(v.y, v.y, s);
    s = fmaf(v.z, v.z, s); s = fmaf(v.w, v.w, s);
    bf4 o;
    o.x = (__bf16)v.x; o.y = (__bf16)v.y; o.z = (__bf16)v.z; o.w = (__bf16)v.w;
    dst[i * 32 + lane] = o;
  }
#pragma unroll
  for (int m = 16; m >= 1; m >>= 1) s += __shfl_xor(s, m, 32);
  if (lane == 0) csq[row] = s;
}

// Fused bf16-WMMA GEMM + running argmin of (||c||^2 - 2 x.c) over this workgroup's K-slice.
__global__ __launch_bounds__(256) void gemm_argmin_kernel(const __bf16* __restrict__ xb,
                                                          const __bf16* __restrict__ cbb,
                                                          const float* __restrict__ csq,
                                                          unsigned long long* __restrict__ keys) {
  extern __shared__ __align__(16) unsigned char smem[];

  const int tid    = threadIdx.x;
  const int lane   = tid & 31;
  const int wave   = tid >> 5;
  const int lane15 = lane & 15;
  const int half   = lane >> 4;
  const int wrow   = (wave >> 1) * 16;   // 4 row groups of 16
  const int colOff = (wave & 1) * 64;    // 2 column halves of 64
  const int mbase  = blockIdx.x * MBLK;
  const int nbase0 = blockIdx.y * NB_COLS;

  // ---- Stage A panel asynchronously: 64 rows x 2048B, 16B chunks, coalesced ----
  {
    const unsigned char* gx = (const unsigned char*)(xb + (size_t)mbase * DD);
#pragma unroll
    for (int i = 0; i < 32; ++i) {
      int c = tid + i * 256;                 // 8192 chunks total; c*16 == row*2048+off*16
      int row = c >> 7, off = c & 127;
      cp16_async(gx + (size_t)c * 16, smem + row * A_ROW_BYTES + off * 16);
    }
  }

  float minv[8];
  int   mini[8];
#pragma unroll
  for (int r = 0; r < 8; ++r) { minv[r] = __builtin_inff(); mini[r] = 0; }

  for (int nb = 0; nb < NB_COLS / NBLK; ++nb) {
    const int nbase = nbase0 + nb * NBLK;
    const unsigned char* gB = (const unsigned char*)(cbb + (size_t)nbase * DD);

    v8f acc[4];
#pragma unroll
    for (int j = 0; j < 4; ++j)
#pragma unroll
      for (int r = 0; r < 8; ++r) acc[j][r] = 0.0f;

    // stage B tile ks=0 into buffer 0: 128 cols x 64B -> 512 x 16B chunks, 2/thread (async)
    {
      int c0 = tid, c1 = tid + 256;
      cp16_async(gB + (size_t)(c0 >> 2) * 2048 + (c0 & 3) * 16,
                 smem + A_BYTES + (c0 >> 2) * B_COL_BYTES + (c0 & 3) * 16);
      cp16_async(gB + (size_t)(c1 >> 2) * 2048 + (c1 & 3) * 16,
                 smem + A_BYTES + (c1 >> 2) * B_COL_BYTES + (c1 & 3) * 16);
    }
    wait_async_all();      // covers A panel (nb==0) and this B tile
    __syncthreads();

    unsigned boff = 0;     // byte offset of current B buffer (0 or B_BUF_BYTES)
    for (int ks = 0; ks < KSTEPS; ++ks) {
      // async prefetch of next tile into the alternate buffer: L2->LDS DMA
      // overlaps with the 4 WMMAs below; no VGPRs consumed.
      const bool pf = (ks < KSTEPS - 1);
      if (pf) {
        int c0 = tid, c1 = tid + 256;
        unsigned char* nxt = smem + A_BYTES + (boff ^ B_BUF_BYTES);
        cp16_async(gB + (size_t)(c0 >> 2) * 2048 + (ks + 1) * 64 + (c0 & 3) * 16,
                   nxt + (c0 >> 2) * B_COL_BYTES + (c0 & 3) * 16);
        cp16_async(gB + (size_t)(c1 >> 2) * 2048 + (ks + 1) * 64 + (c1 & 3) * 16,
                   nxt + (c1 >> 2) * B_COL_BYTES + (c1 & 3) * 16);
      }

      // A fragment: lane holds row (wrow+lane15), K chunks [8h..8h+8) and [16+8h..)
      const unsigned char* arow = smem + (wrow + lane15) * A_ROW_BYTES + ks * 64;
      union { struct { uv4 lo, hi; } q; v16bf v; } af;
      af.q.lo = lds_read16(arow + half * 16);
      af.q.hi = lds_read16(arow + 32 + half * 16);

      const unsigned char* bbase = smem + A_BYTES + boff;
#pragma unroll
      for (int j = 0; j < 4; ++j) {
        const unsigned char* bcol = bbase + (colOff + j * 16 + lane15) * B_COL_BYTES;
        union { struct { uv4 lo, hi; } q; v16bf v; } bfrag;
        bfrag.q.lo = lds_read16(bcol + half * 16);
        bfrag.q.hi = lds_read16(bcol + 32 + half * 16);
        acc[j] = __builtin_amdgcn_wmma_f32_16x16x32_bf16(
            false, af.v, false, bfrag.v, (short)0, acc[j], false, false);
      }

      if (pf) {
        wait_async_all();   // next tile landed in LDS
        __syncthreads();    // all waves done with current buffer + see next buffer
      }
      boff ^= B_BUF_BYTES;
    }

    // running argmin: score = ||c||^2 - 2*cross  (||x||^2 is row-constant, argmin-invariant)
#pragma unroll
    for (int j = 0; j < 4; ++j) {
      int colg = nbase + colOff + j * 16 + lane15;
      float cs = csq[colg];
#pragma unroll
      for (int r = 0; r < 8; ++r) {
        float s = fmaf(-2.0f, acc[j][r], cs);
        if (s < minv[r]) { minv[r] = s; mini[r] = colg; }
      }
    }
    __syncthreads();  // protect B buffers before next nb's staging writes
  }

  // 16-lane argmin reduction (C/D layout: lane half h, VGPR r -> row r + 8h, col lane15)
#pragma unroll
  for (int r = 0; r < 8; ++r) {
    float v = minv[r];
    int ix = mini[r];
#pragma unroll
    for (int m = 1; m < 16; m <<= 1) {
      float ov = __shfl_xor(v, m, 32);
      int oi = __shfl_xor(ix, m, 32);
      if (ov < v || (ov == v && oi < ix)) { v = ov; ix = oi; }
    }
    if (lane15 == 0) {
      unsigned u = __float_as_uint(v);
      u = (u & 0x80000000u) ? ~u : (u | 0x80000000u);  // order-preserving f32->u32 map
      unsigned long long key = ((unsigned long long)u << 32) | (unsigned)ix;
      int grow = mbase + wrow + r + 8 * half;
      atomicMin(&keys[grow], key);  // deterministic cross-split argmin (tie -> smaller index)
    }
  }
}

// out[r,:] = emb[argmin_idx[r], :] + pe[r % T, :]
__global__ __launch_bounds__(256) void gather_kernel(const unsigned long long* __restrict__ keys,
                                                     const float* __restrict__ emb,
                                                     const float* __restrict__ pe,
                                                     float* __restrict__ out) {
  int r = blockIdx.x;
  int t = r & (TT - 1);
  int idx = (int)(unsigned)(keys[r] & 0xFFFFFFFFull);
  const float4* e = (const float4*)(emb + (size_t)idx * DD);
  const float4* p = (const float4*)(pe + (size_t)t * DD);
  float4* o = (float4*)(out + (size_t)r * DD);
  int c = threadIdx.x;                      // 256 threads x float4 = 1024 floats
  float4 ev = e[c], pv = p[c], ov;
  ov.x = ev.x + pv.x; ov.y = ev.y + pv.y;
  ov.z = ev.z + pv.z; ov.w = ev.w + pv.w;
  o[c] = ov;
}

extern "C" void kernel_launch(void* const* d_in, const int* in_sizes, int n_in,
                              void* d_out, int out_size, void* d_ws, size_t ws_size,
                              hipStream_t stream) {
  (void)in_sizes; (void)n_in; (void)out_size; (void)ws_size;
  const float* x   = (const float*)d_in[0];   // [B,T,D]
  const float* cb  = (const float*)d_in[1];   // [K,D]
  const float* emb = (const float*)d_in[2];   // [K,D]
  const float* pe  = (const float*)d_in[3];   // [MAX_LEN,D]
  float* out = (float*)d_out;

  unsigned char* ws = (unsigned char*)d_ws;
  unsigned long long* keys = (unsigned long long*)ws;                 // 64 KB
  float* csq = (float*)(ws + 65536);                                  // 32 KB
  __bf16* xb  = (__bf16*)(ws + 65536 + 32768);                        // 16 MB
  __bf16* cbb = (__bf16*)(ws + 65536 + 32768 + (size_t)MROWS * DD * 2); // 16 MB

  init_keys_kernel<<<(MROWS + 255) / 256, 256, 0, stream>>>(keys);
  convert_x_kernel<<<(MROWS * DD) / 1024, 256, 0, stream>>>(x, xb);
  convert_cb_kernel<<<KCODES / 8, 256, 0, stream>>>(cb, cbb, csq);

  dim3 grid(MROWS / MBLK, NSPLIT);
  gemm_argmin_kernel<<<grid, 256, SMEM_BYTES, stream>>>(xb, cbb, csq, keys);

  gather_kernel<<<MROWS, 256, 0, stream>>>(keys, emb, pe, out);
}
